// VideoEncoder_66554813219297
// MI455X (gfx1250) — matrix operand order repaired and checked by reading, not compiled, not run on hardware
//
#include <hip/hip_runtime.h>
#include <hip/hip_bf16.h>
#include <math.h>

// ---------------- problem constants ----------------
#define B_   32
#define T_   256
#define D_   1024
#define H_   8
#define DH_  128
#define HID_ 256
#define M0_  (B_ * T_)   // 8192 rows of the "token" dimension

typedef __attribute__((ext_vector_type(16))) _Float16 v16h;
typedef __attribute__((ext_vector_type(8)))  float    v8f;

// ---------------- WMMA fragment loaders (f32 memory -> f16 fragments) ----
// A-matrix 16x32 f16: lane row = lane&15;
// lanes 0-15: elems 0..15 = K kb+0..15
// lanes 16-31: elems 0..15 = K kb+16..31 (matches B loaders).
__device__ __forceinline__ v16h load_a_f32(const float* __restrict__ arow,
                                           int kb, int lane) {
    int k0 = kb + ((lane >> 4) << 4);
    v16h f;
#pragma unroll
    for (int i = 0; i < 8; ++i) f[i]     = (_Float16)arow[k0 + i];
#pragma unroll
    for (int i = 0; i < 8; ++i) f[8 + i] = (_Float16)arow[k0 + 8 + i];
    return f;
}

// A fragment from an LDS slab holding K columns 0..31 of the current K-step.
__device__ __forceinline__ v16h load_a_lds(const float* arow, int lane) {
    int k0 = (lane >> 4) << 4;
    v16h f;
#pragma unroll
    for (int i = 0; i < 8; ++i) f[i]     = (_Float16)arow[k0 + i];
#pragma unroll
    for (int i = 0; i < 8; ++i) f[8 + i] = (_Float16)arow[k0 + 8 + i];
    return f;
}

// B-matrix 32x16 f16, B stored row-major K x N (stride ldb):
// lane col = lane&15; lanes 0-15 hold K kb..kb+15, lanes 16-31 hold kb+16..kb+31.
__device__ __forceinline__ v16h load_b_kn(const float* __restrict__ Bw, int ldb,
                                          int kb, int nbase, int lane) {
    int col = nbase + (lane & 15);
    int k0  = kb + ((lane >> 4) << 4);
    const float* p = Bw + (size_t)k0 * ldb + col;
    v16h f;
#pragma unroll
    for (int e = 0; e < 16; ++e) f[e] = (_Float16)p[(size_t)e * ldb];
    return f;
}

// B stored row-major N x K (i.e. transposed weight such as W_ih (3H, D)):
__device__ __forceinline__ v16h load_b_nk(const float* __restrict__ Bw, int ldb,
                                          int kb, int nbase, int lane) {
    int col = nbase + (lane & 15);
    int k0  = kb + ((lane >> 4) << 4);
    const float* p = Bw + (size_t)col * ldb + k0;
    v16h f;
#pragma unroll
    for (int e = 0; e < 16; ++e) f[e] = (_Float16)p[e];
    return f;
}

__device__ __forceinline__ float sigmoidf_(float v) {
    return 1.0f / (1.0f + expf(-v));
}

// ---- CDNA5 async copy: 64 bytes global -> LDS per thread (4x b128) ------
// LDS dest and global src both advance by INST_OFFSET (ISA 10.x async form).
__device__ __forceinline__ void async_copy64(unsigned lds_byte_off,
                                             const float* g) {
    unsigned long long ga = (unsigned long long)(uintptr_t)g;
    asm volatile(
        "global_load_async_to_lds_b128 %0, %1, off\n\t"
        "global_load_async_to_lds_b128 %0, %1, off offset:16\n\t"
        "global_load_async_to_lds_b128 %0, %1, off offset:32\n\t"
        "global_load_async_to_lds_b128 %0, %1, off offset:48"
        :: "v"(lds_byte_off), "v"(ga) : "memory");
}

// ---------------- block-cooperative f16-WMMA GEMM -------------------------
// C = A*B + bias (+ residual).  BT=0: B row-major KxN; BT=1: B row-major NxK.
// Block tile 128(M) x 256(N); 8 waves as 2(M) x 4(N); wave tile 64x64
// (4x4 accumulators -> 16 WMMA per K-step per wave).
// A slab (128x32 f32) double-buffered in LDS via async-to-LDS DMA:
// next slab streams in (ASYNCcnt) while WMMA consumes the current one.
// Grids are exact -> uniform control flow -> EXEC all-ones at every WMMA.
#define APAD 36   // padded LDS row stride (floats) to spread banks
template <int BT>
__global__ __launch_bounds__(256) void gemm_kernel(
    const float* __restrict__ A, const float* __restrict__ Bw,
    const float* __restrict__ bias, const float* __restrict__ resid,
    float* __restrict__ C, int M, int N, int K, int lda, int ldb, int ldc) {
    __shared__ float Atile[2][128 * APAD];   // ~36 KB

    const int wave = threadIdx.x >> 5;
    const int lane = threadIdx.x & 31;
    const int nblk = N >> 8;                 // block-cols along N
    const int mblock = (blockIdx.x / nblk) << 7;
    const int nblock = (blockIdx.x % nblk) << 8;
    const int wm = (wave & 1) << 6;          // wave M offset in block (0/64)
    const int wn = (wave >> 1) << 6;         // wave N offset in block (0..192)

    // staging: 2 threads per slab row, 16 consecutive floats (64 B) each
    const int tid  = threadIdx.x;
    const int srow = tid >> 1;
    const int scol = (tid & 1) << 4;
    const float*   gsrc = A + (size_t)(mblock + srow) * lda + scol;
    const unsigned lds0 =
        (unsigned)(uintptr_t)&Atile[0][srow * APAD + scol];
    const unsigned lds1 =
        (unsigned)(uintptr_t)&Atile[1][srow * APAD + scol];

    v8f acc[4][4] = {};

    async_copy64(lds0, gsrc);                // prologue: slab for kb = 0

    for (int kb = 0, buf = 0; kb < K; kb += 32, buf ^= 1) {
        if (kb + 32 < K) {
            async_copy64(buf ? lds0 : lds1, gsrc + kb + 32);
            // 4 new ops in flight; wait for the 4 belonging to current slab
            asm volatile("s_wait_asynccnt 0x4" ::: "memory");
        } else {
            asm volatile("s_wait_asynccnt 0x0" ::: "memory");
        }
        __syncthreads();                     // slab visible to all waves

        const float* at = &Atile[buf][0];
        v16h af[4];
#pragma unroll
        for (int i = 0; i < 4; ++i)
            af[i] = load_a_lds(at + (wm + (i << 4) + (lane & 15)) * APAD, lane);
#pragma unroll
        for (int j = 0; j < 4; ++j) {
            int nb = nblock + wn + (j << 4);
            v16h bf = BT ? load_b_nk(Bw, ldb, kb, nb, lane)
                         : load_b_kn(Bw, ldb, kb, nb, lane);
#pragma unroll
            for (int i = 0; i < 4; ++i)
                acc[i][j] = __builtin_amdgcn_wmma_f32_16x16x32_f16(
                    false, af[i], false, bf, (short)0, acc[i][j], false, false);
        }
        __syncthreads();                     // done reading before next overwrite
    }

#pragma unroll
    for (int j = 0; j < 4; ++j) {
        int colc = nblock + wn + (j << 4) + (lane & 15);
        float bsv = bias ? bias[colc] : 0.0f;
#pragma unroll
        for (int i = 0; i < 4; ++i) {
            int rbase = mblock + wm + (i << 4) + ((lane >> 4) << 3);
#pragma unroll
            for (int r = 0; r < 8; ++r) {
                size_t o = (size_t)(rbase + r) * ldc + colc;
                float vv = acc[i][j][r] + bsv;
                if (resid) vv += resid[o];
                C[o] = vv;
            }
        }
    }
}

// ---------------- banded attention (window +-3), one wave per (b,h,q) -----
__global__ __launch_bounds__(256) void attn_kernel(
    const float* __restrict__ q, const float* __restrict__ k,
    const float* __restrict__ v, float* __restrict__ o) {
    int wid  = blockIdx.x * 8 + (threadIdx.x >> 5);
    int lane = threadIdx.x & 31;
    int t = wid % T_;
    int hh = (wid / T_) % H_;
    int b = wid / (T_ * H_);
    size_t base = ((size_t)(b * T_ + t)) * D_ + hh * DH_ + lane * 4;
    float4 qv = *(const float4*)(q + base);
    const float scale = 0.08838834764831845f;  // 1/sqrt(128)

    int lo = t - 3; if (lo < 0) lo = 0;
    int hi = t + 3; if (hi > T_ - 1) hi = T_ - 1;
    float sc[7];
    float mx = -1e30f;
    for (int j = lo; j <= hi; ++j) {
        size_t kb = ((size_t)(b * T_ + j)) * D_ + hh * DH_ + lane * 4;
        float4 kv = *(const float4*)(k + kb);
        float p = qv.x * kv.x + qv.y * kv.y + qv.z * kv.z + qv.w * kv.w;
        p += __shfl_xor(p, 16, 32);
        p += __shfl_xor(p, 8, 32);
        p += __shfl_xor(p, 4, 32);
        p += __shfl_xor(p, 2, 32);
        p += __shfl_xor(p, 1, 32);
        p *= scale;
        sc[j - lo] = p;
        if (p > mx) mx = p;
    }
    float sum = 0.0f;
    for (int j = lo; j <= hi; ++j) {
        float e = expf(sc[j - lo] - mx);
        sc[j - lo] = e;
        sum += e;
    }
    float inv = 1.0f / sum;
    float ax = 0, ay = 0, az = 0, aw = 0;
    for (int j = lo; j <= hi; ++j) {
        size_t vb = ((size_t)(b * T_ + j)) * D_ + hh * DH_ + lane * 4;
        float4 vv = *(const float4*)(v + vb);
        float w = sc[j - lo] * inv;
        ax += w * vv.x; ay += w * vv.y; az += w * vv.z; aw += w * vv.w;
    }
    float4 ov; ov.x = ax; ov.y = ay; ov.z = az; ov.w = aw;
    *(float4*)(o + base) = ov;
}

// ---------------- small helpers ----------------
__global__ void lengths_kernel(const int* __restrict__ mask, int* __restrict__ len) {
    int b = threadIdx.x;
    if (b < B_) {
        int s = 0;
        for (int t = 0; t < T_; ++t) s += mask[b * T_ + t];
        len[b] = s;
    }
}

__global__ __launch_bounds__(256) void copy_h_kernel(const float* __restrict__ x,
                                                     float* __restrict__ h) {
    int i = blockIdx.x * 256 + threadIdx.x;   // (b*T+t)*1024 + d
    int d = i & (D_ - 1);
    int row = i >> 10;
    h[i] = x[(size_t)row * 2560 + d];
}

// builds xs (T,B,D) and xs_rev (T,B,D) from h (B,T,D)
__global__ __launch_bounds__(256) void transpose_kernel(
    const float* __restrict__ h, const int* __restrict__ len,
    float* __restrict__ xs, float* __restrict__ xsr) {
    int i = blockIdx.x * 256 + threadIdx.x;   // (b*T+t)*D + d
    int d = i & (D_ - 1);
    int bt = i >> 10;
    int t = bt & (T_ - 1);
    int b = bt >> 8;
    size_t o = ((size_t)(t * B_ + b)) * D_ + d;
    xs[o] = h[i];
    int L = len[b];
    int ri = L - 1 - t; if (ri < 0) ri = 0;
    xsr[o] = h[((size_t)(b * T_ + ri)) * D_ + d];
}

// ---------------- GRU recurrence, WMMA for h @ W_hh^T ---------------------
// grid = 2 blocks (dir), 1024 threads = 32 waves = 2(m) x 16(n) tile grid of
// the 32x256 gate tile; each wave owns aligned r/z/n tiles -> gate math in regs.
__global__ __launch_bounds__(1024) void gru_kernel(
    const float* __restrict__ gx_f, const float* __restrict__ gx_b,
    const float* __restrict__ Whh_f, const float* __restrict__ bhh_f,
    const float* __restrict__ Whh_b, const float* __restrict__ bhh_b,
    float* __restrict__ ys_f, float* __restrict__ ys_b) {
    __shared__ float hst[B_ * HID_];          // 32 KB hidden state
    const int dir = blockIdx.x;
    const float* gx  = dir ? gx_b  : gx_f;
    const float* Whh = dir ? Whh_b : Whh_f;
    const float* bhh = dir ? bhh_b : bhh_f;
    float* ys = dir ? ys_b : ys_f;

    for (int i = threadIdx.x; i < B_ * HID_; i += 1024) hst[i] = 0.0f;
    __syncthreads();

    const int wave  = threadIdx.x >> 5;
    const int lane  = threadIdx.x & 31;
    const int mbase = (wave & 1) << 4;        // batch-row base (0 or 16)
    const int gbase = (wave >> 1) << 4;       // hidden-col base (0..240)
    const int rowA  = mbase + (lane & 15);
    const int colg  = gbase + (lane & 15);
    const int rbase = mbase + ((lane >> 4) << 3);
    const float bh_r = bhh[colg];
    const float bh_z = bhh[HID_ + colg];
    const float bh_n = bhh[2 * HID_ + colg];
    const float* hrow = &hst[rowA * HID_];

    for (int t = 0; t < T_; ++t) {
        v8f cr = {}, cz = {}, cn = {};
        for (int kb = 0; kb < HID_; kb += 32) {
            v16h af = load_a_f32(hrow, kb, lane);
            v16h br = load_b_nk(Whh, HID_, kb, 0 * HID_ + gbase, lane);
            v16h bz = load_b_nk(Whh, HID_, kb, 1 * HID_ + gbase, lane);
            v16h bn = load_b_nk(Whh, HID_, kb, 2 * HID_ + gbase, lane);
            cr = __builtin_amdgcn_wmma_f32_16x16x32_f16(false, af, false, br,
                                                        (short)0, cr, false, false);
            cz = __builtin_amdgcn_wmma_f32_16x16x32_f16(false, af, false, bz,
                                                        (short)0, cz, false, false);
            cn = __builtin_amdgcn_wmma_f32_16x16x32_f16(false, af, false, bn,
                                                        (short)0, cn, false, false);
        }
        float hnew[8];
#pragma unroll
        for (int r = 0; r < 8; ++r) {
            int b = rbase + r;
            const float* g = gx + ((size_t)(t * B_ + b)) * (3 * HID_) + colg;
            float rr = sigmoidf_(g[0]        + cr[r] + bh_r);
            float zz = sigmoidf_(g[HID_]     + cz[r] + bh_z);
            float nn = tanhf(g[2 * HID_] + rr * (cn[r] + bh_n));
            hnew[r] = (1.0f - zz) * nn + zz * hst[b * HID_ + colg];
        }
        __syncthreads();                       // all reads of old h done
#pragma unroll
        for (int r = 0; r < 8; ++r) {
            int b = rbase + r;
            hst[b * HID_ + colg] = hnew[r];
            ys[((size_t)(t * B_ + b)) * HID_ + colg] = hnew[r];
        }
        __syncthreads();                       // new h visible for next step
    }
}

// ---------------- final combine: out = [ys_f ; ys_b(rev)] * valid + x_face -
__global__ __launch_bounds__(256) void combine_kernel(
    const float* __restrict__ ysf, const float* __restrict__ ysb,
    const float* __restrict__ xf, const int* __restrict__ len,
    float* __restrict__ out) {
    int i = blockIdx.x * 256 + threadIdx.x;   // (b*T + t)*512 + c
    int c = i & 511;
    int bt = i >> 9;
    int t = bt & (T_ - 1);
    int b = bt >> 8;
    int L = len[b];
    float val = 0.0f;
    if (t < L) {
        if (c < HID_) {
            val = ysf[((size_t)(t * B_ + b)) * HID_ + c];
        } else {
            int ri = L - 1 - t; if (ri < 0) ri = 0;
            val = ysb[((size_t)(ri * B_ + b)) * HID_ + (c - HID_)];
        }
    }
    out[i] = val + xf[i];
}

// ---------------- host launch ----------------
extern "C" void kernel_launch(void* const* d_in, const int* in_sizes, int n_in,
                              void* d_out, int out_size, void* d_ws, size_t ws_size,
                              hipStream_t stream) {
    (void)in_sizes; (void)n_in; (void)out_size; (void)ws_size;
    const float* x      = (const float*)d_in[0];
    const int*   mask   = (const int*)d_in[1];
    const float* W_tran = (const float*)d_in[2];
    const float* b_tran = (const float*)d_in[3];
    const float* Wq     = (const float*)d_in[4];
    const float* bq     = (const float*)d_in[5];
    const float* Wk     = (const float*)d_in[6];
    const float* bk     = (const float*)d_in[7];
    const float* Wv     = (const float*)d_in[8];
    const float* bv     = (const float*)d_in[9];
    const float* Wo     = (const float*)d_in[10];
    const float* bo     = (const float*)d_in[11];
    const float* W_ih_f = (const float*)d_in[12];
    const float* W_hh_f = (const float*)d_in[13];
    const float* b_ih_f = (const float*)d_in[14];
    const float* b_hh_f = (const float*)d_in[15];
    const float* W_ih_b = (const float*)d_in[16];
    const float* W_hh_b = (const float*)d_in[17];
    const float* b_ih_b = (const float*)d_in[18];
    const float* b_hh_b = (const float*)d_in[19];

    const size_t MEG = 1024 * 1024;
    float* W   = (float*)d_ws;
    float* h   = W;              // 8M  (B,T,D) residual stream
    float* qb  = W + 8 * MEG;    // 8M
    float* kb_ = W + 16 * MEG;   // 8M
    float* vb  = W + 24 * MEG;   // 8M
    float* ao  = W + 32 * MEG;   // 8M  attention output
    float* xf  = W + 40 * MEG;   // 4M  x_face
    // GRU phase reuses q/k/v/ao regions:
    float* xs  = qb;             // 8M  (T,B,D)
    float* xsr = kb_;            // 8M  (T,B,D) reversed
    float* gxf = vb;             // 6M  (T,B,768)
    float* gxb = ao;             // 6M
    float* ysf = vb + 6 * MEG;   // 2M  (T,B,256)
    float* ysb = ao + 6 * MEG;   // 2M
    int*   lenp = (int*)(W + 44 * MEG);

    // sequence lengths + hidden-stream init
    lengths_kernel<<<1, 32, 0, stream>>>(mask, lenp);
    copy_h_kernel<<<(M0_ * D_) / 256, 256, 0, stream>>>(x, h);

    // x_face = x[:,:,1024:] @ W_tran + b_tran
    gemm_kernel<0><<<(M0_ / 128) * (512 / 256), 256, 0, stream>>>(
        x + 1024, W_tran, b_tran, nullptr, xf, M0_, 512, 1536, 2560, 512, 512);

    // two transformer layers
    for (int l = 0; l < 2; ++l) {
        const float* wq = Wq + (size_t)l * D_ * D_;
        const float* wk = Wk + (size_t)l * D_ * D_;
        const float* wv = Wv + (size_t)l * D_ * D_;
        const float* wo = Wo + (size_t)l * D_ * D_;
        int grid = (M0_ / 128) * (D_ / 256);
        gemm_kernel<0><<<grid, 256, 0, stream>>>(h, wq, bq + l * D_, nullptr, qb,
                                                 M0_, D_, D_, D_, D_, D_);
        gemm_kernel<0><<<grid, 256, 0, stream>>>(h, wk, bk + l * D_, nullptr, kb_,
                                                 M0_, D_, D_, D_, D_, D_);
        gemm_kernel<0><<<grid, 256, 0, stream>>>(h, wv, bv + l * D_, nullptr, vb,
                                                 M0_, D_, D_, D_, D_, D_);
        attn_kernel<<<(B_ * H_ * T_) / 8, 256, 0, stream>>>(qb, kb_, vb, ao);
        // h = h + ao @ Wo + bo  (in-place residual: each element reads only itself)
        gemm_kernel<0><<<grid, 256, 0, stream>>>(ao, wo, bo + l * D_, h, h,
                                                 M0_, D_, D_, D_, D_, D_);
    }

    // time-major transpose + reversed copy
    transpose_kernel<<<(M0_ * D_) / 256, 256, 0, stream>>>(h, lenp, xs, xsr);

    // GRU input projections: gx = xs @ W_ih^T + b_ih   (W_ih is (3H, D) -> BT=1)
    {
        int grid = (M0_ / 128) * (3 * HID_ / 256);
        gemm_kernel<1><<<grid, 256, 0, stream>>>(xs, W_ih_f, b_ih_f, nullptr, gxf,
                                                 M0_, 3 * HID_, D_, D_, D_, 3 * HID_);
        gemm_kernel<1><<<grid, 256, 0, stream>>>(xsr, W_ih_b, b_ih_b, nullptr, gxb,
                                                 M0_, 3 * HID_, D_, D_, D_, 3 * HID_);
    }

    // sequential bidirectional GRU (one workgroup per direction, WMMA inside)
    gru_kernel<<<2, 1024, 0, stream>>>(gxf, gxb, W_hh_f, b_hh_f, W_hh_b, b_hh_b,
                                       ysf, ysb);

    // out = concat(ys_f, ys_b reversed) * valid + x_face
    combine_kernel<<<(B_ * T_ * 512) / 256, 256, 0, stream>>>(ysf, ysb, xf, lenp,
                                                              (float*)d_out);
}
